// chebLSTMCell_14663018348905
// MI455X (gfx1250) — compile-verified, matrix operand order checked by reading, not run on hardware
//
#include <hip/hip_runtime.h>
#include <hip/hip_bf16.h>

// ---------- types ----------
typedef __attribute__((ext_vector_type(4)))  unsigned int u4;
typedef __attribute__((ext_vector_type(4)))  float        f4;
typedef __attribute__((ext_vector_type(16))) __bf16       v16bf;
typedef __attribute__((ext_vector_type(8)))  float        v8f;
typedef int gv_i4 __attribute__((vector_size(4 * sizeof(int))));  // matches builtin param

struct U4x2 { u4 lo; u4 hi; };

__device__ __forceinline__ v16bf frag_from(u4 lo, u4 hi) {
    U4x2 t; t.lo = lo; t.hi = hi;
    return __builtin_bit_cast(v16bf, t);
}
__device__ __forceinline__ unsigned short f2bf(float f) {
    __bf16 h = (__bf16)f;
    return __builtin_bit_cast(unsigned short, h);
}

#define NTOK 4096
#define FC   128      // concat feature dim (F_IN + H)
#define NB   8        // batch
#define HD   64       // hidden
#define KC   32       // k-chunk per WMMA
#define MT   64       // L-rows per block in cheb_gemm
#define LDSB_STR 40   // 32 + 8 pad (bf16 elems); 80B rows keep 16B align

// ---- CDNA5 async global->LDS path (guarded; falls back to sync staging) ----
#if defined(__has_builtin)
#if __has_builtin(__builtin_amdgcn_global_load_async_to_lds_b128) && \
    __has_builtin(__builtin_amdgcn_s_wait_asynccnt)
#define USE_ASYNC 1
#endif
#endif
#ifndef USE_ASYNC
#define USE_ASYNC 0
#endif

#define AS1 __attribute__((address_space(1)))
#define AS3 __attribute__((address_space(3)))

// ---------------------------------------------------------------------------
// K0a: Zt[b][f][n] = bf16( f<64 ? x[b][n][f] : h[b][n][f-64] )  (LDS-tiled)
// ---------------------------------------------------------------------------
__global__ __launch_bounds__(256) void pack_zt(const float* __restrict__ x,
                                               const float* __restrict__ h,
                                               unsigned short* __restrict__ Zt) {
    __shared__ unsigned short tile[FC][34];   // [feat][token], +2 pad
    const int b  = blockIdx.y;
    const int t0 = blockIdx.x * 32;
    const int tid = threadIdx.x;

    const int tok = t0 + (tid >> 3);
    const int f0  = (tid & 7) * 16;
    const float* src = (f0 < 64)
        ? x + ((size_t)b * NTOK + tok) * 64 + f0
        : h + ((size_t)b * NTOK + tok) * 64 + (f0 - 64);
    f4 a = ((const f4*)src)[0];
    f4 bq = ((const f4*)src)[1];
    f4 c = ((const f4*)src)[2];
    f4 d = ((const f4*)src)[3];
    const int tl = tok - t0;
    tile[f0+ 0][tl]=f2bf(a.x);  tile[f0+ 1][tl]=f2bf(a.y);
    tile[f0+ 2][tl]=f2bf(a.z);  tile[f0+ 3][tl]=f2bf(a.w);
    tile[f0+ 4][tl]=f2bf(bq.x); tile[f0+ 5][tl]=f2bf(bq.y);
    tile[f0+ 6][tl]=f2bf(bq.z); tile[f0+ 7][tl]=f2bf(bq.w);
    tile[f0+ 8][tl]=f2bf(c.x);  tile[f0+ 9][tl]=f2bf(c.y);
    tile[f0+10][tl]=f2bf(c.z);  tile[f0+11][tl]=f2bf(c.w);
    tile[f0+12][tl]=f2bf(d.x);  tile[f0+13][tl]=f2bf(d.y);
    tile[f0+14][tl]=f2bf(d.z);  tile[f0+15][tl]=f2bf(d.w);
    __syncthreads();

    const int f  = tid >> 1;
    const int tk = (tid & 1) * 16;
    union { unsigned short s[16]; u4 q[2]; } pk;
#pragma unroll
    for (int i = 0; i < 16; ++i) pk.s[i] = tile[f][tk + i];
    unsigned short* dst = Zt + ((size_t)b * FC + f) * NTOK + t0 + tk;
    ((u4*)dst)[0] = pk.q[0];
    ((u4*)dst)[1] = pk.q[1];
}

// ---------------------------------------------------------------------------
// K0b: Wct[k][o][f] = bf16( f<64 ? W1[k][f][o] : W2[k][f-64][o] ); bias = b1+b2
// ---------------------------------------------------------------------------
__global__ void pack_w(const float* __restrict__ W1, const float* __restrict__ W2,
                       const float* __restrict__ b1, const float* __restrict__ b2,
                       unsigned short* __restrict__ Wct, float* __restrict__ biasc) {
    const int idx = blockIdx.x * 256 + threadIdx.x;   // 3*256*128 = 98304 total
    if (idx < 3 * 256 * 128) {
        const int f = idx & 127;
        const int o = (idx >> 7) & 255;
        const int k = idx >> 15;
        const float w = (f < 64) ? W1[((size_t)k * 64 + f) * 256 + o]
                                 : W2[((size_t)k * 64 + (f - 64)) * 256 + o];
        Wct[idx] = f2bf(w);
    }
    if (blockIdx.x == 0 && threadIdx.x < 256)
        biasc[threadIdx.x] = b1[threadIdx.x] + b2[threadIdx.x];
}

// ---------------------------------------------------------------------------
// K1/K2: C = L @ B  (per batch, M=4096, N=128, K=4096), bf16 WMMA, fp32 acc.
// Block: 64 threads (2 waves), 64 rows. Wave: 32 rows x 128 cols
// (two A fragments per wave -> each B fragment feeds two WMMAs).
// Double-buffered LDS; B chunks staged via async global->LDS when available.
// ---------------------------------------------------------------------------
__global__ __launch_bounds__(64) void cheb_gemm(
    const float* __restrict__ Lg,              // [B,4096,4096] f32
    const unsigned short* __restrict__ Bt,     // [B,128,4096] bf16 (feature-major)
    unsigned short* __restrict__ Ct,           // [B,128,4096] bf16 or null
    unsigned short* __restrict__ Crm,          // [B,4096,128] bf16
    const float* __restrict__ xin, const float* __restrict__ hin,
    int pass)
{
    __shared__ unsigned short ldsB[2][FC * LDSB_STR];
    const int b    = blockIdx.y;
    const int m0   = blockIdx.x * MT;
    const int tid  = threadIdx.x;
    const int wave = tid >> 5;
    const int lane = tid & 31;
    const int l15  = lane & 15;
    const int lhalf = lane >> 4;
    const int mw   = m0 + wave * 32;           // wave covers 32 rows

    const float* Arow0 = Lg + (((size_t)b * NTOK + mw) + l15) * NTOK;
    const float* Arow1 = Arow0 + (size_t)16 * NTOK;
    const unsigned short* Btb = Bt + (size_t)b * FC * NTOK;

    // staging: each thread owns feature rows 2*tid and 2*tid+1 (64B each)
    const unsigned short* srow0 = Btb + (size_t)(2 * tid) * NTOK;
    const unsigned short* srow1 = srow0 + NTOK;

    v8f accA[8] = {};
    v8f accB[8] = {};

    // ---- stage helper (async if available, else VGPR copy) ----
    auto stage = [&](int buf, int k0) {
        unsigned short* d0 = &ldsB[buf][(2 * tid) * LDSB_STR];
        unsigned short* d1 = d0 + LDSB_STR;
        const unsigned short* s0 = srow0 + k0;
        const unsigned short* s1 = srow1 + k0;
#if USE_ASYNC
        __builtin_amdgcn_global_load_async_to_lds_b128((AS1 gv_i4*)(s0 +  0), (AS3 gv_i4*)(d0 +  0), 0, 0);
        __builtin_amdgcn_global_load_async_to_lds_b128((AS1 gv_i4*)(s0 +  8), (AS3 gv_i4*)(d0 +  8), 0, 0);
        __builtin_amdgcn_global_load_async_to_lds_b128((AS1 gv_i4*)(s0 + 16), (AS3 gv_i4*)(d0 + 16), 0, 0);
        __builtin_amdgcn_global_load_async_to_lds_b128((AS1 gv_i4*)(s0 + 24), (AS3 gv_i4*)(d0 + 24), 0, 0);
        __builtin_amdgcn_global_load_async_to_lds_b128((AS1 gv_i4*)(s1 +  0), (AS3 gv_i4*)(d1 +  0), 0, 0);
        __builtin_amdgcn_global_load_async_to_lds_b128((AS1 gv_i4*)(s1 +  8), (AS3 gv_i4*)(d1 +  8), 0, 0);
        __builtin_amdgcn_global_load_async_to_lds_b128((AS1 gv_i4*)(s1 + 16), (AS3 gv_i4*)(d1 + 16), 0, 0);
        __builtin_amdgcn_global_load_async_to_lds_b128((AS1 gv_i4*)(s1 + 24), (AS3 gv_i4*)(d1 + 24), 0, 0);
#else
        ((u4*)d0)[0] = ((const u4*)s0)[0]; ((u4*)d0)[1] = ((const u4*)s0)[1];
        ((u4*)d0)[2] = ((const u4*)s0)[2]; ((u4*)d0)[3] = ((const u4*)s0)[3];
        ((u4*)d1)[0] = ((const u4*)s1)[0]; ((u4*)d1)[1] = ((const u4*)s1)[1];
        ((u4*)d1)[2] = ((const u4*)s1)[2]; ((u4*)d1)[3] = ((const u4*)s1)[3];
#endif
    };

    stage(0, 0);                                // prologue: chunk 0 -> buf 0

    int buf = 0;
    for (int k0 = 0; k0 < NTOK; k0 += KC, buf ^= 1) {
#if USE_ASYNC
        __builtin_amdgcn_s_wait_asynccnt(0);    // my async writes to ldsB[buf] done
#endif
        __syncthreads();                        // all staging of ldsB[buf] visible

        if (k0 + KC < NTOK)
            stage(buf ^ 1, k0 + KC);            // overlaps with compute below

        // prefetch streamed L rows two chunks ahead
        __builtin_prefetch(Arow0 + k0 + 2 * KC, 0, 1);
        __builtin_prefetch(Arow1 + k0 + 2 * KC, 0, 1);

        // ---- A fragments: L fp32 -> bf16, documented 16-bit A layout ----
        // lanes 0-15: K {0..7, 16..23}; lanes 16-31: K {8..15, 24..31}
        const float* a0 = Arow0 + k0 + lhalf * 8;
        const float* a1 = Arow0 + k0 + 16 + lhalf * 8;
        const float* a2 = Arow1 + k0 + lhalf * 8;
        const float* a3 = Arow1 + k0 + 16 + lhalf * 8;
        f4 q0 = ((const f4*)a0)[0];
        f4 q1 = ((const f4*)a0)[1];
        f4 q2 = ((const f4*)a1)[0];
        f4 q3 = ((const f4*)a1)[1];
        f4 r0 = ((const f4*)a2)[0];
        f4 r1 = ((const f4*)a2)[1];
        f4 r2 = ((const f4*)a3)[0];
        f4 r3 = ((const f4*)a3)[1];
        v16bf afr0, afr1;
        afr0[0]=(__bf16)q0.x; afr0[1]=(__bf16)q0.y; afr0[2]=(__bf16)q0.z; afr0[3]=(__bf16)q0.w;
        afr0[4]=(__bf16)q1.x; afr0[5]=(__bf16)q1.y; afr0[6]=(__bf16)q1.z; afr0[7]=(__bf16)q1.w;
        afr0[8]=(__bf16)q2.x; afr0[9]=(__bf16)q2.y; afr0[10]=(__bf16)q2.z; afr0[11]=(__bf16)q2.w;
        afr0[12]=(__bf16)q3.x; afr0[13]=(__bf16)q3.y; afr0[14]=(__bf16)q3.z; afr0[15]=(__bf16)q3.w;
        afr1[0]=(__bf16)r0.x; afr1[1]=(__bf16)r0.y; afr1[2]=(__bf16)r0.z; afr1[3]=(__bf16)r0.w;
        afr1[4]=(__bf16)r1.x; afr1[5]=(__bf16)r1.y; afr1[6]=(__bf16)r1.z; afr1[7]=(__bf16)r1.w;
        afr1[8]=(__bf16)r2.x; afr1[9]=(__bf16)r2.y; afr1[10]=(__bf16)r2.z; afr1[11]=(__bf16)r2.w;
        afr1[12]=(__bf16)r3.x; afr1[13]=(__bf16)r3.y; afr1[14]=(__bf16)r3.z; afr1[15]=(__bf16)r3.w;

        // ---- 8 column tiles; each B fragment feeds two WMMAs ----
#pragma unroll
        for (int j = 0; j < 8; ++j) {
            const int col = j * 16 + l15;
            const unsigned short* bp = &ldsB[buf][col * LDSB_STR + lhalf * 8];
            u4 blo = *(const u4*)bp;          // K {0..7}+half
            u4 bhi = *(const u4*)(bp + 16);   // K {16..23}+half
            v16bf bfr = frag_from(blo, bhi);
            accA[j] = __builtin_amdgcn_wmma_f32_16x16x32_bf16(
                false, afr0, false, bfr, (short)0, accA[j], false, false);
            accB[j] = __builtin_amdgcn_wmma_f32_16x16x32_bf16(
                false, afr1, false, bfr, (short)0, accB[j], false, false);
        }
        __syncthreads();                        // done reading ldsB[buf]
    }

    // ---- epilogue: C layout -> global (two 16-row tiles per wave) ----
#pragma unroll
    for (int half = 0; half < 2; ++half) {
        const int token_base = mw + half * 16 + lhalf * 8;
#pragma unroll
        for (int j = 0; j < 8; ++j) {
            const int ncol = j * 16 + l15;
#pragma unroll
            for (int i = 0; i < 8; ++i) {
                float v = half ? accB[j][i] : accA[j][i];
                const int token = token_base + i;
                if (pass == 2) {
                    const float z = (ncol < 64)
                        ? xin[((size_t)b * NTOK + token) * 64 + ncol]
                        : hin[((size_t)b * NTOK + token) * 64 + (ncol - 64)];
                    v = 2.0f * v - z;
                }
                const unsigned short hv = f2bf(v);
                Crm[((size_t)b * NTOK + token) * FC + ncol] = hv;
                if (Ct) Ct[((size_t)b * FC + ncol) * NTOK + token] = hv;
            }
        }
    }
}

// ---------------------------------------------------------------------------
// K3: combined = Z@Wc0 + T1@Wc1 + T2@Wc2 + bias -> LSTM gates -> h_next,c_next
// Block: 256 threads (8 waves), 128 tokens; wave: 16 tokens x 256 cols.
// ---------------------------------------------------------------------------
__global__ __launch_bounds__(256) void proj_lstm(
    const float* __restrict__ x, const float* __restrict__ h,
    const float* __restrict__ ccur,
    const unsigned short* __restrict__ T1rm,   // [B,4096,128] bf16
    const unsigned short* __restrict__ T2rm,   // [B,4096,128] bf16
    const unsigned short* __restrict__ Wct,    // [3,256,128] bf16 (out-major)
    const float* __restrict__ biasc,           // [256] f32
    float* __restrict__ out)                   // h_next | c_next
{
    const int b    = blockIdx.y;
    const int t0   = blockIdx.x * 128;
    const int wave = threadIdx.x >> 5;
    const int lane = threadIdx.x & 31;
    const int l15  = lane & 15;
    const int lhalf = lane >> 4;
    const int tw   = t0 + wave * 16;

    v8f acc[16] = {};

    for (int c = 0; c < 12; ++c) {
        const int km  = c >> 2;          // 0=Z (from x/h), 1=T1, 2=T2
        const int kf0 = (c & 3) * 32;

        // ---- A fragment (token rows, contiguous feature K) ----
        v16bf afr;
        const int tokenA = tw + l15;
        if (km == 0) {
            const int fb0 = kf0 + lhalf * 8;
            const int fb1 = kf0 + 16 + lhalf * 8;
            const float* p0 = (fb0 < 64)
                ? x + ((size_t)b * NTOK + tokenA) * 64 + fb0
                : h + ((size_t)b * NTOK + tokenA) * 64 + (fb0 - 64);
            const float* p1 = (fb1 < 64)
                ? x + ((size_t)b * NTOK + tokenA) * 64 + fb1
                : h + ((size_t)b * NTOK + tokenA) * 64 + (fb1 - 64);
            f4 q0 = ((const f4*)p0)[0];
            f4 q1 = ((const f4*)p0)[1];
            f4 q2 = ((const f4*)p1)[0];
            f4 q3 = ((const f4*)p1)[1];
            afr[0]=(__bf16)q0.x; afr[1]=(__bf16)q0.y; afr[2]=(__bf16)q0.z; afr[3]=(__bf16)q0.w;
            afr[4]=(__bf16)q1.x; afr[5]=(__bf16)q1.y; afr[6]=(__bf16)q1.z; afr[7]=(__bf16)q1.w;
            afr[8]=(__bf16)q2.x; afr[9]=(__bf16)q2.y; afr[10]=(__bf16)q2.z; afr[11]=(__bf16)q2.w;
            afr[12]=(__bf16)q3.x; afr[13]=(__bf16)q3.y; afr[14]=(__bf16)q3.z; afr[15]=(__bf16)q3.w;
        } else {
            const unsigned short* T = (km == 1) ? T1rm : T2rm;
            const unsigned short* p =
                T + ((size_t)b * NTOK + tokenA) * FC + kf0 + lhalf * 8;
            afr = frag_from(((const u4*)p)[0], *(const u4*)(p + 16));
        }

        // ---- B fragments from Wct (out-major, contiguous K) + WMMA ----
        const unsigned short* Wk = Wct + (size_t)km * 256 * FC;
#pragma unroll
        for (int j = 0; j < 16; ++j) {
            const int col = j * 16 + l15;
            const unsigned short* p = Wk + (size_t)col * FC + kf0 + lhalf * 8;
            acc[j] = __builtin_amdgcn_wmma_f32_16x16x32_bf16(
                false, afr, false,
                frag_from(((const u4*)p)[0], *(const u4*)(p + 16)),
                (short)0, acc[j], false, false);
        }
    }

    // ---- fused bias + LSTM gates ----
    const size_t HOFF = (size_t)NB * NTOK * HD;   // h_next | c_next split
#pragma unroll
    for (int jc = 0; jc < 4; ++jc) {
        const int col = jc * 16 + l15;            // hidden column 0..63
        const float bi = biasc[col];
        const float bfv = biasc[64 + col];
        const float bo = biasc[128 + col];
        const float bg = biasc[192 + col];
#pragma unroll
        for (int i = 0; i < 8; ++i) {
            const int token = tw + lhalf * 8 + i;
            const size_t base = ((size_t)b * NTOK + token) * HD + col;
            const float vi = acc[jc][i]      + bi;
            const float vf = acc[jc + 4][i]  + bfv;
            const float vo = acc[jc + 8][i]  + bo;
            const float vg = acc[jc + 12][i] + bg;
            const float ig = 1.0f / (1.0f + __expf(-vi));
            const float fg = 1.0f / (1.0f + __expf(-vf));
            const float og = 1.0f / (1.0f + __expf(-vo));
            const float gg = tanhf(vg);
            const float cn = fg * ccur[base] + ig * gg;
            const float hn = og * tanhf(cn);
            out[base]        = hn;
            out[HOFF + base] = cn;
        }
    }
}

// ---------------------------------------------------------------------------
extern "C" void kernel_launch(void* const* d_in, const int* in_sizes, int n_in,
                              void* d_out, int out_size, void* d_ws, size_t ws_size,
                              hipStream_t stream) {
    const float* x  = (const float*)d_in[0];   // [8,4096,64]
    const float* L  = (const float*)d_in[1];   // [8,4096,4096]
    const float* h  = (const float*)d_in[2];   // [8,4096,64]
    const float* c  = (const float*)d_in[3];   // [8,4096,64]
    const float* W1 = (const float*)d_in[4];   // [3,64,256]
    const float* b1 = (const float*)d_in[5];   // [256]
    const float* W2 = (const float*)d_in[6];   // [3,64,256]
    const float* b2 = (const float*)d_in[7];   // [256]

    const size_t nT = (size_t)NB * FC * NTOK;  // 4,194,304 bf16 elems per tensor
    unsigned short* Zt   = (unsigned short*)d_ws;   // [B,128,4096] feature-major
    unsigned short* T1t  = Zt   + nT;               // [B,128,4096] feature-major
    unsigned short* T1rm = T1t  + nT;               // [B,4096,128] row-major
    unsigned short* T2rm = T1rm + nT;               // [B,4096,128] row-major
    unsigned short* Wct  = T2rm + nT;               // [3,256,128]
    float*          biasc = (float*)(Wct + 3 * 256 * FC);

    pack_zt <<<dim3(NTOK / 32, NB), 256, 0, stream>>>(x, h, Zt);
    pack_w  <<<dim3(384),           256, 0, stream>>>(W1, W2, b1, b2, Wct, biasc);
    // T1 = L @ Z
    cheb_gemm<<<dim3(NTOK / MT, NB), 64, 0, stream>>>(L, Zt,  T1t,    T1rm, x, h, 1);
    // T2 = 2 * L @ T1 - Z
    cheb_gemm<<<dim3(NTOK / MT, NB), 64, 0, stream>>>(L, T1t, nullptr, T2rm, x, h, 2);
    // combined -> gates -> (h_next, c_next)
    proj_lstm<<<dim3(NTOK / 128, NB), 256, 0, stream>>>(x, h, c, T1rm, T2rm, Wct,
                                                        biasc, (float*)d_out);
}
